// DeformableConv2d_66520453480941
// MI455X (gfx1250) — matrix operand order repaired and checked by reading, not compiled, not run on hardware
//
#include <hip/hip_runtime.h>
#include <hip/hip_bf16.h>

// ---------------------------------------------------------------------------
// Deformable Conv2d for MI455X (gfx1250, wave32, WMMA)
//   conv1 (offsets, N=18 pad 32)  -> bf16 WMMA implicit GEMM (halo-padded A)
//   deform bilinear sample        -> VALU gather, writes halo-padded NHWC bf16
//   conv2 (64->64)                -> bf16 WMMA implicit GEMM + LDS-transposed
//                                    coalesced b128 stores
// ---------------------------------------------------------------------------

#define B_      8
#define CIN     64
#define COUT    64
#define H_      128
#define W_      128
#define HW      (H_ * W_)          // 16384
#define P_TOTAL (B_ * HW)          // 131072 pixels
#define KK      9
#define KDIM    (CIN * KK)         // 576
#define KCHUNKS (KDIM / 32)        // 18
#define NTILES2 4                  // conv2: 64 out channels = 4 x 16
#define NTILES1 2                  // conv1: 18 out channels padded to 32
#define TILES   (P_TOTAL / 16)     // 8192 pixel tiles (16 pixels along x)

// zero-halo padded NHWC staging (bf16): [B][H+2][W+2][64]
#define HP      (H_ + 2)           // 130
#define WP      (W_ + 2)           // 130
#define RS      (WP * 64)          // padded row stride (elements)
#define IS      ((size_t)HP * RS)  // padded image stride (elements)

typedef __attribute__((ext_vector_type(16))) __bf16 v16bf;
typedef __attribute__((ext_vector_type(8)))  float  v8f;

union Frag {
    v16bf v;
    uint4 q[2];
};

// ---------------------------------------------------------------------------
// Pack OIHW fp32 weights into per-lane WMMA B fragments (bf16).
// Fragment layout: [kc][nt][lane][e], e = 0..15 contiguous bf16 (32B/lane).
// B matrix (32x16) lane mapping: N = nt*16 + (lane&15), K = (lane>>4)*8 + (e&7) + (e>>3)*16,
// global K = kc*32 + Klocal, K ordered (ky,kx) outer, ci inner.
// ---------------------------------------------------------------------------
__global__ void pack_weights(const float* __restrict__ w, int nch, int ntiles,
                             __bf16* __restrict__ frag) {
    int idx = blockIdx.x * blockDim.x + threadIdx.x;
    int total = KCHUNKS * ntiles * 32 * 16;
    if (idx >= total) return;
    int e    = idx & 15;
    int lane = (idx >> 4) & 31;
    int rest = idx >> 9;
    int nt   = rest % ntiles;
    int kc   = rest / ntiles;

    int n  = nt * 16 + (lane & 15);
    int kk = kc >> 1;
    int ci = ((kc & 1) << 5) + ((lane >> 4) << 3) + (e & 7) + ((e >> 3) << 4);

    float val = 0.0f;
    if (n < nch) val = w[(n * CIN + ci) * KK + kk];   // OIHW, kk = ky*3+kx
    frag[idx] = (__bf16)val;
}

// NCHW fp32 -> halo-padded NHWC bf16 (A-operand staging for conv1)
__global__ void nchw_to_nhwc_bf16(const float* __restrict__ x,
                                  __bf16* __restrict__ xbf) {
    int idx = blockIdx.x * blockDim.x + threadIdx.x;   // over P_TOTAL*CIN
    if (idx >= P_TOTAL * CIN) return;
    int c = idx & 63;
    int p = idx >> 6;
    int b = p >> 14;
    int rem = p & 16383;                               // y*W + x
    int y = rem >> 7;
    int xc = rem & 127;
    xbf[(size_t)b * IS + (size_t)(y + 1) * RS + (xc + 1) * 64 + c] =
        (__bf16)x[((size_t)b * CIN + c) * HW + rem];
}

// Compile-time-unrolled N-tile WMMA chain so reuse_a is an ICE immediate.
template <int NT, int NTILES>
__device__ __forceinline__ void wmma_steps(const __bf16* __restrict__ bf_kc,
                                           const Frag& a, v8f* acc) {
    if constexpr (NT < NTILES) {
        Frag bw;
        const uint4* bq = (const uint4*)(bf_kc + NT * 512);
        bw.q[0] = bq[0];
        bw.q[1] = bq[1];
        // reuse_a: back-to-back identical WMMAs share the A fragment
        acc[NT] = __builtin_amdgcn_wmma_f32_16x16x32_bf16(
            false, a.v, false, bw.v, (short)0, acc[NT],
            /*reuse_a=*/(NT > 0), /*reuse_b=*/false);
        wmma_steps<NT + 1, NTILES>(bf_kc, a, acc);
    }
}

// ---------------------------------------------------------------------------
// Core implicit-GEMM tile: one wave computes a 16(pixel) x NTILES*16 tile.
// A fragment per lane: row M = lane&15, kbase = (lane>>4)*8; elements
// e<8 -> K = kbase+e, e>=8 -> K = 16+kbase+(e-8)  (ISA 16-bit A layout),
// i.e. two contiguous 16-byte loads per lane from the halo-padded NHWC
// buffer — unconditional, no edge predication.
// ---------------------------------------------------------------------------
template <int NTILES>
__device__ __forceinline__ void conv_gemm_tile(
    const __bf16* __restrict__ A_pad,    // [B][HP][WP][64] bf16, zero halo
    const __bf16* __restrict__ bfrag,    // [kc][nt][lane][16] bf16
    int tile, int lane, v8f acc[NTILES]) {

    int b   = tile >> 10;                // 1024 tiles per batch image
    int rem = tile & 1023;
    int y   = rem >> 3;
    int x0  = (rem & 7) << 4;            // 8 tiles of 16 pixels per row
    int m   = lane & 15;                 // A-matrix row this lane feeds
    int kb  = (lane >> 4) << 3;          // 0 or 8 (K sub-offset)

    // halo coords: sample (y+ky, x0+m+kx), ky,kx in 0..2, always in range
    const __bf16* base =
        A_pad + (size_t)b * IS + (size_t)y * RS + (x0 + m) * 64 + kb;

    for (int kc = 0; kc < KCHUNKS; ++kc) {
        int kk = kc >> 1;
        int ky = kk / 3, kx = kk - ky * 3;
        int chalf = (kc & 1) << 5;       // which 32-channel half of this tap

        const __bf16* src = base + (size_t)ky * RS + kx * 64 + chalf;
        Frag a;
        a.q[0] = *(const uint4*)(src);        // K = kbase .. kbase+7
        a.q[1] = *(const uint4*)(src + 16);   // K = 16+kbase .. +7

        const __bf16* bf_kc = bfrag + (size_t)kc * (NTILES * 512) + lane * 16;
        wmma_steps<0, NTILES>(bf_kc, a, acc);
    }
}

// conv1: offsets = conv3x3(x) + b_off; store pixel-major fp32 [p][18]
// (stores are already coalesced: 16 lanes write 16 consecutive floats)
__global__ void conv1_wmma(const __bf16* __restrict__ xbf,
                           const __bf16* __restrict__ bfrag,
                           const float* __restrict__ bias,
                           float* __restrict__ offs) {
    int wave = threadIdx.x >> 5;
    int lane = threadIdx.x & 31;
    int tile = blockIdx.x * 4 + wave;

    v8f acc[NTILES1] = {};
    conv_gemm_tile<NTILES1>(xbf, bfrag, tile, lane, acc);

    int n_lo = lane & 15;
    int m_hi = (lane >> 4) << 3;         // C/D layout: lanes 16-31 hold M+8
#pragma unroll
    for (int nt = 0; nt < NTILES1; ++nt) {
        int co = nt * 16 + n_lo;
        if (co < 2 * KK) {
            float bv = bias[co];
#pragma unroll
            for (int r = 0; r < 8; ++r) {
                int p = tile * 16 + (r + m_hi);
                offs[(size_t)p * (2 * KK) + co] = acc[nt][r] + bv;
            }
        }
    }
}

// Deformable bilinear sampling summed over the 9 taps -> halo-padded NHWC bf16
__global__ void deform_sample(const float* __restrict__ x,      // NCHW fp32
                              const float* __restrict__ offs,   // [p][18]
                              __bf16* __restrict__ sampled) {   // padded NHWC
    int p  = blockIdx.x * blockDim.x + threadIdx.x;
    int cg = blockIdx.y;                 // 16-channel group 0..3
    if (p >= P_TOTAL) return;

    int b   = p >> 14;
    int rem = p & 16383;
    int yc  = rem >> 7;
    int xc  = rem & 127;

    const float* xb = x + ((size_t)b * CIN + cg * 16) * HW;
    const float* op = offs + (size_t)p * (2 * KK);

    float acc[16];
#pragma unroll
    for (int c = 0; c < 16; ++c) acc[c] = 0.0f;

    for (int kk = 0; kk < KK; ++kk) {
        float xs = op[2 * kk]     + (float)xc;
        float ys = op[2 * kk + 1] + (float)yc;
        float x0f = floorf(xs), y0f = floorf(ys);
        float wx1 = xs - x0f, wx0 = 1.0f - wx1;
        float wy1 = ys - y0f, wy0 = 1.0f - wy1;
        int x0i = (int)x0f, y0i = (int)y0f;
#pragma unroll
        for (int t = 0; t < 4; ++t) {
            int xi = x0i + (t & 1);
            int yi = y0i + (t >> 1);
            float w = ((t & 1) ? wx1 : wx0) * ((t >> 1) ? wy1 : wy0);
            if (xi >= 0 && xi < W_ && yi >= 0 && yi < H_) {   // zeros pad
                const float* src = xb + yi * W_ + xi;
#pragma unroll
                for (int c = 0; c < 16; ++c)
                    acc[c] = fmaf(w, src[(size_t)c * HW], acc[c]);
            }
        }
    }

    __bf16* dst = sampled + (size_t)b * IS + (size_t)(yc + 1) * RS +
                  (xc + 1) * 64 + cg * 16;
#pragma unroll
    for (int c = 0; c < 16; ++c) dst[c] = (__bf16)acc[c];
}

// conv2: out = conv3x3(sampled) + b_reg; NCHW fp32 output.
// Each wave's 16(pixel) x 64(co) fp32 tile is transposed through LDS
// ([x][co] layout) so the block emits coalesced global_store_b128.
__global__ void conv2_wmma(const __bf16* __restrict__ sampled,
                           const __bf16* __restrict__ bfrag,
                           const float* __restrict__ bias,
                           float* __restrict__ out) {
    __shared__ float lds[4 * 16 * 64];   // 16 KB: 4 waves x 16 px x 64 co

    int wave = threadIdx.x >> 5;
    int lane = threadIdx.x & 31;
    int tile = blockIdx.x * 4 + wave;

    v8f acc[NTILES2] = {};
    conv_gemm_tile<NTILES2>(sampled, bfrag, tile, lane, acc);

    int n_lo = lane & 15;
    int m_hi = (lane >> 4) << 3;         // C/D layout: lanes 16-31 hold M+8
    float* ws = lds + wave * (16 * 64);
#pragma unroll
    for (int nt = 0; nt < NTILES2; ++nt) {
        int co = nt * 16 + n_lo;
        float bv = bias[co];
#pragma unroll
        for (int r = 0; r < 8; ++r)
            ws[(r + m_hi) * 64 + co] = acc[nt][r] + bv;   // [x][co]
    }
    __syncthreads();

    // 1024 chunks of 4 consecutive x per block; co varies fastest across
    // lanes -> conflict-free LDS reads, b128 global stores.
#pragma unroll
    for (int i = 0; i < 8; ++i) {
        int ch = i * 128 + threadIdx.x;
        int co = ch & 63;
        int xg = (ch >> 6) & 3;
        int tl = ch >> 8;                 // tile_local 0..3
        int t2 = blockIdx.x * 4 + tl;
        int b  = t2 >> 10;
        int rm = t2 & 1023;
        int y  = rm >> 3;
        int x0 = (rm & 7) << 4;

        const float* src = lds + tl * (16 * 64) + (xg * 4) * 64 + co;
        float4 v;
        v.x = src[0];
        v.y = src[64];
        v.z = src[128];
        v.w = src[192];
        *(float4*)&out[(((size_t)b * COUT + co) * H_ + y) * W_ + x0 + xg * 4] = v;
    }
}

// ---------------------------------------------------------------------------
extern "C" void kernel_launch(void* const* d_in, const int* in_sizes, int n_in,
                              void* d_out, int out_size, void* d_ws,
                              size_t ws_size, hipStream_t stream) {
    const float* x     = (const float*)d_in[0];   // (8,64,128,128)
    const float* w_off = (const float*)d_in[1];   // (18,64,3,3)
    const float* b_off = (const float*)d_in[2];   // (18,)
    const float* w_reg = (const float*)d_in[3];   // (64,64,3,3)
    const float* b_reg = (const float*)d_in[4];   // (64,)
    float* out = (float*)d_out;                   // (8,64,128,128)

    // workspace carve-up (256B aligned)
    size_t cur = 0;
    auto carve = [&](size_t bytes) {
        void* p = (char*)d_ws + cur;
        cur += (bytes + 255) & ~(size_t)255;
        return p;
    };
    size_t pad_bytes = (size_t)B_ * IS * sizeof(__bf16);               // 17.3 MB
    __bf16* xbf     = (__bf16*)carve(pad_bytes);
    __bf16* sampled = (__bf16*)carve(pad_bytes);
    float*  offs    = (float*)carve((size_t)P_TOTAL * 2 * KK * 4);     // 9 MB
    __bf16* bfrag2  = (__bf16*)carve((size_t)KCHUNKS * NTILES2 * 512 * 2);
    __bf16* bfrag1  = (__bf16*)carve((size_t)KCHUNKS * NTILES1 * 512 * 2);

    // 0) zero the halo-padded staging buffers (graph-capturable memset)
    (void)hipMemsetAsync(xbf, 0, pad_bytes, stream);
    (void)hipMemsetAsync(sampled, 0, pad_bytes, stream);

    // 1) weight fragment packing (tiny)
    {
        int tot1 = KCHUNKS * NTILES1 * 512;
        pack_weights<<<(tot1 + 255) / 256, 256, 0, stream>>>(w_off, 2 * KK,
                                                             NTILES1, bfrag1);
        int tot2 = KCHUNKS * NTILES2 * 512;
        pack_weights<<<(tot2 + 255) / 256, 256, 0, stream>>>(w_reg, COUT,
                                                             NTILES2, bfrag2);
    }
    // 2) stage x as halo-padded NHWC bf16
    {
        int tot = P_TOTAL * CIN;
        nchw_to_nhwc_bf16<<<(tot + 255) / 256, 256, 0, stream>>>(x, xbf);
    }
    // 3) offsets conv (WMMA)
    conv1_wmma<<<TILES / 4, 128, 0, stream>>>(xbf, bfrag1, b_off, offs);
    // 4) deformable bilinear gather
    {
        dim3 grid(P_TOTAL / 256, 4, 1);
        deform_sample<<<grid, 256, 0, stream>>>(x, offs, sampled);
    }
    // 5) main 64->64 conv (WMMA)
    conv2_wmma<<<TILES / 4, 128, 0, stream>>>(sampled, bfrag2, b_reg, out);
}
